// RNNBlock_19258633355340
// MI455X (gfx1250) — matrix operand order repaired and checked by reading, not compiled, not run on hardware
//
#include <hip/hip_runtime.h>

// ---- problem constants (match reference) ----
#define B_   65536
#define T_   79
#define F_   8
#define C_   256
#define DW_  256
#define HPAD 264   // 256 + 8 bf16 padding -> row stride 528B (16B aligned, bank-skewed)

typedef __attribute__((ext_vector_type(16))) __bf16 v16bf;
typedef __attribute__((ext_vector_type(8)))  __bf16 v8bf;
typedef __attribute__((ext_vector_type(8)))  float  v8f;

static __device__ __forceinline__ __bf16 f2bf(float f) {
    // round-to-nearest-even fp32 -> bf16
    unsigned u = __builtin_bit_cast(unsigned, f);
    unsigned r = u + 0x7FFFu + ((u >> 16) & 1u);
    unsigned short h = (unsigned short)(r >> 16);
    return __builtin_bit_cast(__bf16, h);
}

__global__ __launch_bounds__(512) void rnn_wmma_fused(
    const float* __restrict__ x,   // (B, T, F)
    const float* __restrict__ W,   // (F, C)
    const float* __restrict__ U,   // (C, C)
    const float* __restrict__ b,   // (C)
    const float* __restrict__ W1,  // (C, DW)
    const float* __restrict__ b1,  // (DW)
    float* __restrict__ out)       // (B, DW)
{
    __shared__ __align__(16) __bf16 hbuf[2][16][HPAD];   // ping-pong hidden state (bf16)

    const int tid  = threadIdx.x;
    const int lane = tid & 31;
    const int wave = tid >> 5;            // 0..15 : this wave's 16-column N-tile
    const int l15  = lane & 15;
    const bool hi  = lane >= 16;
    const int n    = wave * 16 + l15;     // global output column owned by this lane
    const int base = blockIdx.x * 16;     // batch-row base of this workgroup

    // ---------------------------------------------------------------
    // Pre-load B-fragments for Wcat = [W ; zeros ; U]  (288 x 256), K-blocks of 32.
    // B-fragment layout (16-bit, 32x16): element e of lane L holds
    //   K = kblock + (L>=16 ? 16 : 0) + e,  N = 16*wave + (L&15)
    // ---------------------------------------------------------------
    const int kb = hi ? 16 : 0;
    v16bf bw[9];
    #pragma unroll
    for (int e = 0; e < 16; ++e) {                 // frag 0: K 0..7 = W, K 8..31 = 0
        const int kg = kb + e;
        bw[0][e] = f2bf(kg < F_ ? W[kg * C_ + n] : 0.0f);
    }
    #pragma unroll
    for (int s = 1; s <= 8; ++s)                   // frags 1..8: U rows 0..255
        #pragma unroll
        for (int e = 0; e < 16; ++e)
            bw[s][e] = f2bf(U[(32 * (s - 1) + kb + e) * C_ + n]);

    const float bias = b[n];

    // ---- h0 = 0
    for (int i = tid; i < 16 * HPAD; i += 512)
        (&hbuf[0][0][0])[i] = f2bf(0.0f);
    __syncthreads();

    // A-fragment 0 skeleton for x_t: elements 8..15 (K8..15) and all hi-lane
    // elements (K16..31 >= F) are permanently zero -> init ONCE.
    v16bf a0;
    #pragma unroll
    for (int e = 0; e < 16; ++e) a0[e] = f2bf(0.0f);

    // ---------------------------------------------------------------
    // Recurrence: t = T-1 .. 0  (go_backwards)
    // Per step: 9 x v_wmma_f32_16x16x32_bf16 over K = [x|pad|h] = 288
    // ---------------------------------------------------------------
    int p = 0;
    const int cA = hi ? 16 : 0;  // A-fragment per-lane K sub-offset
    for (int step = 0; step < T_; ++step) {
        const int t = T_ - 1 - step;

        v8f acc;
        #pragma unroll
        for (int r = 0; r < 8; ++r) acc[r] = bias;

        // overwrite only the 8 live feature elements of a0 (lanes 0-15)
        if (!hi) {
            const float* xr = x + ((size_t)(base + l15) * T_ + t) * F_;
            #pragma unroll
            for (int e = 0; e < F_; ++e) a0[e] = f2bf(xr[e]);
        }
        acc = __builtin_amdgcn_wmma_f32_16x16x32_bf16(false, a0, false, bw[0],
                                                      (short)0, acc, false, false);

        // A-fragments 1..8: h_prev from LDS (already bf16, 16B vector loads)
        #pragma unroll
        for (int s = 1; s <= 8; ++s) {
            const int c0 = 32 * (s - 1);
            const v8bf lo  = *(const v8bf*)&hbuf[p][l15][c0 + cA];
            const v8bf hi8 = *(const v8bf*)&hbuf[p][l15][c0 + 8 + cA];
            v16bf a;
            #pragma unroll
            for (int e = 0; e < 8; ++e) { a[e] = lo[e]; a[8 + e] = hi8[e]; }
            acc = __builtin_amdgcn_wmma_f32_16x16x32_bf16(false, a, false, bw[s],
                                                          (short)0, acc, false, false);
        }

        // relu (v_max_num_f32) -> bf16 -> h_next
        // (C/D layout: VGPR r holds M = r, +8 for hi lanes)
        #pragma unroll
        for (int r = 0; r < 8; ++r) {
            const int m = (hi ? 8 : 0) + r;
            hbuf[1 - p][m][n] = f2bf(__builtin_fmaxf(acc[r], 0.0f));
        }
        __syncthreads();
        p ^= 1;
    }

    // ---------------------------------------------------------------
    // Final dense: out = relu(h @ W1 + b1), K = 256 -> 8 WMMAs
    // ---------------------------------------------------------------
    v16bf bw1[8];
    #pragma unroll
    for (int s = 0; s < 8; ++s)
        #pragma unroll
        for (int e = 0; e < 16; ++e)
            bw1[s][e] = f2bf(W1[(32 * s + kb + e) * DW_ + n]);

    v8f acc;
    const float bias1 = b1[n];
    #pragma unroll
    for (int r = 0; r < 8; ++r) acc[r] = bias1;

    #pragma unroll
    for (int s = 0; s < 8; ++s) {
        const int c0 = 32 * s;
        const v8bf lo  = *(const v8bf*)&hbuf[p][l15][c0 + cA];
        const v8bf hi8 = *(const v8bf*)&hbuf[p][l15][c0 + 8 + cA];
        v16bf a;
        #pragma unroll
        for (int e = 0; e < 8; ++e) { a[e] = lo[e]; a[8 + e] = hi8[e]; }
        acc = __builtin_amdgcn_wmma_f32_16x16x32_bf16(false, a, false, bw1[s],
                                                      (short)0, acc, false, false);
    }

    // streaming (non-temporal) stores: write-once output, don't pollute L2
    #pragma unroll
    for (int r = 0; r < 8; ++r) {
        const int m = (hi ? 8 : 0) + r;
        const float v = __builtin_fmaxf(acc[r], 0.0f);
        __builtin_nontemporal_store(v, &out[(size_t)(base + m) * DW_ + n]);
    }
}

extern "C" void kernel_launch(void* const* d_in, const int* in_sizes, int n_in,
                              void* d_out, int out_size, void* d_ws, size_t ws_size,
                              hipStream_t stream) {
    (void)in_sizes; (void)n_in; (void)d_ws; (void)ws_size; (void)out_size;
    const float* x  = (const float*)d_in[0];
    const float* W  = (const float*)d_in[1];
    const float* U  = (const float*)d_in[2];
    const float* b  = (const float*)d_in[3];
    const float* W1 = (const float*)d_in[4];
    const float* b1 = (const float*)d_in[5];
    float* out = (float*)d_out;

    dim3 grid(B_ / 16);   // 4096 workgroups, each covers 16 batch rows x 256 cols
    dim3 block(512);      // 16 waves (wave32): one N-tile per wave
    rnn_wmma_fused<<<grid, block, 0, stream>>>(x, W, U, b, W1, b1, out);
}
